// PointCnnEncoderPool_19026705121656
// MI455X (gfx1250) — matrix operand order repaired and checked by reading, not compiled, not run on hardware
//
#include <hip/hip_runtime.h>
#include <hip/hip_bf16.h>

// ---------------- constants ----------------
#define BATCH 4
#define N0    4096
#define KNN_T 128
#define KMAX  20
#define FPS_T 1024

typedef _Float16 half16 __attribute__((ext_vector_type(16)));
typedef _Float16 half8v __attribute__((ext_vector_type(8)));
typedef float    float8 __attribute__((ext_vector_type(8)));

__device__ __forceinline__ float act_apply(float v, int act) {
  if (act == 1) return v > 0.f ? v : (expf(v) - 1.f);   // ELU (alpha=1)
  if (act == 2) return v > 0.f ? v : 0.f;               // ReLU
  return v;
}

// ---------------- kNN (self-inclusive, ascending distance, stable ties) ----------------
__global__ __launch_bounds__(KNN_T) void knn_kernel(
    const float* __restrict__ pos, int n, int K, int* __restrict__ idx_out)
{
  __shared__ float sx[KNN_T], sy[KNN_T], sz[KNN_T];
  int bpb = n / KNN_T;
  int b = blockIdx.x / bpb;
  int q = (blockIdx.x % bpb) * KNN_T + threadIdx.x;
  const float* pb = pos + (size_t)b * n * 3;
  float qx = pb[q*3+0], qy = pb[q*3+1], qz = pb[q*3+2];
  float bd[KMAX]; int bi[KMAX];
  for (int k = 0; k < K; ++k) { bd[k] = 3.4e38f; bi[k] = 0; }
  for (int t0 = 0; t0 < n; t0 += KNN_T) {
    int j = t0 + threadIdx.x;
    sx[threadIdx.x] = pb[j*3+0];
    sy[threadIdx.x] = pb[j*3+1];
    sz[threadIdx.x] = pb[j*3+2];
    __syncthreads();
    for (int u = 0; u < KNN_T; ++u) {
      float dx = sx[u]-qx, dy = sy[u]-qy, dz = sz[u]-qz;
      float d = dx*dx + dy*dy + dz*dz;
      if (d < bd[K-1]) {                       // strict <: earlier index wins ties
        int p = K - 1;
        while (p > 0 && d < bd[p-1]) { bd[p] = bd[p-1]; bi[p] = bi[p-1]; --p; }
        bd[p] = d; bi[p] = t0 + u;
      }
    }
    __syncthreads();
  }
  int* op = idx_out + ((size_t)b * n + q) * K;
  for (int k = 0; k < K; ++k) op[k] = bi[k];
}

// ---------------- pdiff[b,i,k,:] = pos[idx]-pos[i] ----------------
__global__ void pdiff_kernel(const float* __restrict__ pos, const int* __restrict__ idx,
                             int n, int K, float* __restrict__ pdiff, int total)
{
  int t = blockIdx.x * blockDim.x + threadIdx.x;
  if (t >= total) return;
  int i = (t / K) % n;
  int b =  t / (K * n);
  const float* pb = pos + (size_t)b * n * 3;
  int j = idx[t];
  pdiff[t*3+0] = pb[j*3+0] - pb[i*3+0];
  pdiff[t*3+1] = pb[j*3+1] - pb[i*3+1];
  pdiff[t*3+2] = pb[j*3+2] - pb[i*3+2];
}

// ---------------- small scalar linear: Y = act(X[M,Kc]@W[Kc,N]+b) ----------------
__global__ void linear_act_kernel(const float* __restrict__ X, const float* __restrict__ W,
                                  const float* __restrict__ bias, float* __restrict__ Y,
                                  int N, int Kc, int act, int total)
{
  int stride = gridDim.x * blockDim.x;
  for (int t = blockIdx.x * blockDim.x + threadIdx.x; t < total; t += stride) {
    int col = t % N; int row = t / N;
    float acc = bias[col];
    const float* xr = X + (size_t)row * Kc;
    for (int k = 0; k < Kc; ++k) acc += xr[k] * W[k * N + col];
    Y[t] = act_apply(acc, act);
  }
}

// ---------------- BatchNorm (training mode): stats then apply ----------------
__global__ __launch_bounds__(256) void bn_stats_kernel(
    const float* __restrict__ X, int M, int C, float* __restrict__ mean, float* __restrict__ rsig)
{
  int c = blockIdx.x, tid = threadIdx.x;
  float s = 0.f, s2 = 0.f;
  for (int r = tid; r < M; r += 256) { float v = X[(size_t)r * C + c]; s += v; s2 += v * v; }
  __shared__ float sh[256], sh2[256];
  sh[tid] = s; sh2[tid] = s2; __syncthreads();
  for (int off = 128; off > 0; off >>= 1) {
    if (tid < off) { sh[tid] += sh[tid+off]; sh2[tid] += sh2[tid+off]; }
    __syncthreads();
  }
  if (tid == 0) {
    float mu = sh[0] / (float)M;
    float var = sh2[0] / (float)M - mu * mu;
    if (var < 0.f) var = 0.f;
    mean[c] = mu;
    rsig[c] = rsqrtf(var + 1e-5f);
  }
}

__global__ void bn_apply_kernel(float* __restrict__ X, const float* __restrict__ mean,
                                const float* __restrict__ rsig, const float* __restrict__ g,
                                const float* __restrict__ be, int C, int total)
{
  int stride = gridDim.x * blockDim.x;
  for (int t = blockIdx.x * blockDim.x + threadIdx.x; t < total; t += stride) {
    int c = t % C;
    X[t] = (X[t] - mean[c]) * rsig[c] * g[c] + be[c];
  }
}

// ---------------- operand packing for WMMA ----------------
// A: f32 [M,Kc] -> f16 [M,Kcp], zero-padded K (Kcp multiple of 32)
__global__ void packA_kernel(const float* __restrict__ X, _Float16* __restrict__ Xh,
                             int Kc, int Kcp, int total)
{
  int t = blockIdx.x * blockDim.x + threadIdx.x;
  if (t >= total) return;
  int k = t % Kcp; int row = t / Kcp;
  Xh[t] = (k < Kc) ? (_Float16)X[(size_t)row * Kc + k] : (_Float16)0.f;
}
// B: f32 [Kc,N] -> f16 panel-transposed Bp[kstep][col][t] = W[kstep*32+t][col]
__global__ void packB_kernel(const float* __restrict__ W, _Float16* __restrict__ Bp,
                             int Kc, int N, int total)
{
  int t = blockIdx.x * blockDim.x + threadIdx.x;
  if (t >= total) return;
  int tt  = t & 31;
  int col = (t >> 5) % N;
  int ks  = t / (32 * N);
  int k = ks * 32 + tt;
  Bp[t] = (k < Kc) ? (_Float16)W[(size_t)k * N + col] : (_Float16)0.f;
}

// ---------------- WMMA GEMM: Y = act(Xh[M,Kcp] @ W + bias), f16 in / f32 acc ----------------
// One wave per 16x16 tile, 8 waves per block. M,N multiples of 16; Kcp multiple of 32.
// All fragment loads are contiguous, 16B-aligned, branch-free.
__global__ __launch_bounds__(256) void gemm_wmma_kernel(
    const _Float16* __restrict__ Xh, const _Float16* __restrict__ Bp,
    const float* __restrict__ bias, float* __restrict__ Y,
    int M, int N, int Kcp, int act)
{
  int ntn = N >> 4;
  int ntiles = (M >> 4) * ntn;
  int wid = threadIdx.x >> 5;
  int tile = blockIdx.x * 8 + wid;
  if (tile >= ntiles) return;            // whole wave exits uniformly (EXEC all-1s for WMMA)
  int tm = (tile / ntn) << 4;
  int tn = (tile % ntn) << 4;
  int lane = threadIdx.x & 31;
  int hh  = lane >> 4;                    // half of wave
  int l16 = lane & 15;
  float8 acc = {0.f,0.f,0.f,0.f,0.f,0.f,0.f,0.f};
  // A: lane row = tm+l16; per step needs f16 [k0+hh*8 .. +7] and [k0+16+hh*8 .. +7]
  const _Float16* xr = Xh + (size_t)(tm + l16) * Kcp + hh * 8;
  // B: lane col = tn+l16; per step needs f16 at panel offset [hh*8 .. +7] and [16+hh*8 .. +7]
  const _Float16* br = Bp + ((size_t)(tn + l16)) * 32 + hh * 8;
  int nks = Kcp >> 5;
  #pragma unroll 2
  for (int ks = 0; ks < nks; ++ks) {
    const _Float16* xa = xr + ks * 32;
    half8v alo = *(const half8v*)(xa);
    half8v ahi = *(const half8v*)(xa + 16);
    half16 a = __builtin_shufflevector(alo, ahi, 0,1,2,3,4,5,6,7,8,9,10,11,12,13,14,15);
    const _Float16* xb = br + (size_t)ks * N * 32;
    half8v blo = *(const half8v*)(xb);
    half8v bhi = *(const half8v*)(xb + 16);
    half16 b = __builtin_shufflevector(blo, bhi, 0,1,2,3,4,5,6,7,8,9,10,11,12,13,14,15);
    acc = __builtin_amdgcn_wmma_f32_16x16x32_f16(false, a, false, b, (short)0, acc, false, false);
  }
  int col = tn + l16;
  float bv = bias[col];
  #pragma unroll
  for (int r = 0; r < 8; ++r) {          // C/D: VGPR r -> row r + 8*hh, col = l16
    int row = tm + r + 8 * hh;
    Y[(size_t)row * N + col] = act_apply(acc[r] + bv, act);
  }
}

// ---------------- X-transform refinement: out[m,g,j] = act(sum_k t[m,g,k]*cw[g,j,k] + cb[g,j]) ----------------
__global__ void einsum_cw_kernel(const float* __restrict__ Tin, const float* __restrict__ cw,
                                 const float* __restrict__ cb, float* __restrict__ Tout,
                                 int K, int act, int total)
{
  int stride = gridDim.x * blockDim.x;
  for (int t = blockIdx.x * blockDim.x + threadIdx.x; t < total; t += stride) {
    int j = t % K;
    int g = (t / K) % K;
    int m = t / (K * K);
    const float* tr = Tin + ((size_t)m * K + g) * K;
    const float* wr = cw + ((size_t)g * K + j) * K;
    float acc = cb[g * K + j];
    for (int k = 0; k < K; ++k) acc += tr[k] * wr[k];
    Tout[t] = act_apply(acc, act);
  }
}

// ---------------- fused: xs gather -> xt = xs@t -> depthwise -> fcin[m, c*dm+d] ----------------
__global__ __launch_bounds__(256) void xtrans_dw_kernel(
    const float* __restrict__ X, int Cin, const float* __restrict__ H, int Cd,
    const int* __restrict__ idx, const float* __restrict__ T,
    const float* __restrict__ dw, const float* __restrict__ db,
    float* __restrict__ out, int n, int K, int dm)
{
  extern __shared__ float st[];                 // K*K transform for this point
  int pt = blockIdx.x;                          // 0 .. B*n-1
  int b = pt / n;
  int KK = K * K;
  const float* tp = T + (size_t)pt * KK;
  for (int u = threadIdx.x; u < KK; u += 256) st[u] = tp[u];
  __syncthreads();
  int C = Cin + Cd, CD = C * dm;
  for (int od = threadIdx.x; od < CD; od += 256) {
    int c = od / dm;
    float xs[KMAX];
    if (c < Cd) {
      for (int k2 = 0; k2 < K; ++k2) xs[k2] = H[((size_t)pt * K + k2) * Cd + c];
    } else {
      int cc = c - Cd;
      for (int k2 = 0; k2 < K; ++k2) {
        int j = idx[(size_t)pt * K + k2];
        xs[k2] = X[((size_t)b * n + j) * Cin + cc];
      }
    }
    float acc = db[od];
    const float* dwr = dw + (size_t)od * K;     // dw[C,dm,K] flat: (c*dm+d)*K
    for (int k = 0; k < K; ++k) {
      float xt = 0.f;
      for (int k2 = 0; k2 < K; ++k2) xt += xs[k2] * st[k2 * K + k];
      acc += xt * dwr[k];
    }
    out[(size_t)pt * CD + od] = acc;
  }
}

// ---------------- farthest point sampling: one block per batch ----------------
__global__ __launch_bounds__(FPS_T) void fps_kernel(
    const float* __restrict__ pos, int n, int m, int* __restrict__ out)
{
  int b = blockIdx.x, tid = threadIdx.x;
  const float* pb = pos + (size_t)b * n * 3;
  int ppt = n / FPS_T;                           // <= 4
  float px[4], py[4], pz[4], dist[4];
  float c0x = pb[0], c0y = pb[1], c0z = pb[2];
  for (int i = 0; i < ppt; ++i) {
    int j = i * FPS_T + tid;
    px[i] = pb[j*3+0]; py[i] = pb[j*3+1]; pz[i] = pb[j*3+2];
    float dx = px[i]-c0x, dy = py[i]-c0y, dz = pz[i]-c0z;
    dist[i] = dx*dx + dy*dy + dz*dz;
  }
  if (tid == 0) out[b * m] = 0;
  __shared__ float rv[FPS_T];
  __shared__ int   ri[FPS_T];
  __shared__ float cur[3];
  for (int s = 1; s < m; ++s) {
    float best = -1.f; int bj = 0x7fffffff;
    for (int i = 0; i < ppt; ++i) {
      int j = i * FPS_T + tid;
      if (dist[i] > best) { best = dist[i]; bj = j; }   // strict >: first max wins
    }
    rv[tid] = best; ri[tid] = bj;
    __syncthreads();
    for (int off = FPS_T / 2; off > 0; off >>= 1) {
      if (tid < off) {
        if (rv[tid+off] > rv[tid] || (rv[tid+off] == rv[tid] && ri[tid+off] < ri[tid])) {
          rv[tid] = rv[tid+off]; ri[tid] = ri[tid+off];
        }
      }
      __syncthreads();
    }
    int nxt = ri[0];
    if (tid == (nxt & (FPS_T - 1))) {
      int sl = nxt / FPS_T;
      cur[0] = px[sl]; cur[1] = py[sl]; cur[2] = pz[sl];
      out[b * m + s] = nxt;
    }
    __syncthreads();
    float cx = cur[0], cy = cur[1], cz = cur[2];
    for (int i = 0; i < ppt; ++i) {
      float dx = px[i]-cx, dy = py[i]-cy, dz = pz[i]-cz;
      float d = dx*dx + dy*dy + dz*dz;
      dist[i] = fminf(dist[i], d);
    }
    __syncthreads();
  }
}

// ---------------- gather rows per batch: dst[b,j,:] = src[b, fidx[b,j], :] ----------------
__global__ void gather_kernel(const float* __restrict__ src, const int* __restrict__ fidx,
                              float* __restrict__ dst, int n, int m, int C, int total)
{
  int t = blockIdx.x * blockDim.x + threadIdx.x;
  if (t >= total) return;
  int c = t % C;
  int j = (t / C) % m;
  int b = t / (C * m);
  int s = fidx[b * m + j];
  dst[t] = src[((size_t)b * n + s) * C + c];
}

__global__ void copy_kernel(float* __restrict__ dst, const float* __restrict__ src, int count)
{
  int t = blockIdx.x * blockDim.x + threadIdx.x;
  if (t < count) dst[t] = src[t];
}

__global__ void iota_batch_kernel(float* __restrict__ dst, int per, int count)
{
  int t = blockIdx.x * blockDim.x + threadIdx.x;
  if (t < count) dst[t] = (float)(t / per);
}

// ================= host side =================
enum {
  PW1=0, PB1, PG1, PBE1, PW2, PB2, PG2, PBE2,
  PLW, PLB, PLG, PLBE, PCW1, PCB1, PCG1, PCBE1,
  PCW2, PCB2, PCG2, PCBE2, PDW, PDB, PFW, PFB
};

struct Ws {
  int   *idx, *fpsidx;
  float *pdiff, *hA, *hB, *tA, *tB, *fcin, *featA, *featB, *pos2, *pos3, *mean, *rsig;
  _Float16 *xh, *bp;
};

static inline int cdiv(int a, int b) { return (a + b - 1) / b; }

static void launch_gemm(const float* X, const float* W, const float* bias, float* Y,
                        int M, int N, int Kc, int act, const Ws& w, hipStream_t s)
{
  int Kcp = (Kc + 31) & ~31;
  packA_kernel<<<cdiv(M * Kcp, 256), 256, 0, s>>>(X, w.xh, Kc, Kcp, M * Kcp);
  packB_kernel<<<cdiv(Kcp * N, 256), 256, 0, s>>>(W, w.bp, Kc, N, Kcp * N);
  int tiles = (M >> 4) * (N >> 4);
  gemm_wmma_kernel<<<cdiv(tiles, 8), 256, 0, s>>>(w.xh, w.bp, bias, Y, M, N, Kcp, act);
}

static void run_xconv(const float* x, int Cin, const float* pos, int n,
                      int K, int Cd, int Cout, const float* const* P,
                      float* out, const Ws& w, hipStream_t s)
{
  int C  = Cin + Cd;
  int dm = (Cout + C - 1) / C;
  int M1 = BATCH * n * K;      // rows for mlp1
  int M2 = BATCH * n;          // points
  int KK = K * K;

  // kNN + relative positions
  knn_kernel<<<BATCH * (n / KNN_T), KNN_T, 0, s>>>(pos, n, K, w.idx);
  pdiff_kernel<<<cdiv(M1, 256), 256, 0, s>>>(pos, w.idx, n, K, w.pdiff, M1);

  // mlp1: lift pdiff to Cd features (BN with batch stats)
  linear_act_kernel<<<cdiv(M1 * Cd, 256), 256, 0, s>>>(w.pdiff, P[PW1], P[PB1], w.hA, Cd, 3, 1, M1 * Cd);
  bn_stats_kernel<<<Cd, 256, 0, s>>>(w.hA, M1, Cd, w.mean, w.rsig);
  bn_apply_kernel<<<cdiv(M1 * Cd, 256), 256, 0, s>>>(w.hA, w.mean, w.rsig, P[PG1], P[PBE1], Cd, M1 * Cd);
  linear_act_kernel<<<cdiv(M1 * Cd, 256), 256, 0, s>>>(w.hA, P[PW2], P[PB2], w.hB, Cd, Cd, 1, M1 * Cd);
  bn_stats_kernel<<<Cd, 256, 0, s>>>(w.hB, M1, Cd, w.mean, w.rsig);
  bn_apply_kernel<<<cdiv(M1 * Cd, 256), 256, 0, s>>>(w.hB, w.mean, w.rsig, P[PG2], P[PBE2], Cd, M1 * Cd);

  // mlp2: X-transform [B*n, 3K] @ lw[3K, K^2]  (WMMA) + ELU, then BN
  launch_gemm(w.pdiff, P[PLW], P[PLB], w.tA, M2, KK, 3 * K, 1, w, s);
  bn_stats_kernel<<<KK, 256, 0, s>>>(w.tA, M2, KK, w.mean, w.rsig);
  bn_apply_kernel<<<cdiv(M2 * KK, 256), 256, 0, s>>>(w.tA, w.mean, w.rsig, P[PLG], P[PLBE], KK, M2 * KK);

  einsum_cw_kernel<<<cdiv(M2 * KK, 256), 256, 0, s>>>(w.tA, P[PCW1], P[PCB1], w.tB, K, 1, M2 * KK);
  bn_stats_kernel<<<KK, 256, 0, s>>>(w.tB, M2, KK, w.mean, w.rsig);
  bn_apply_kernel<<<cdiv(M2 * KK, 256), 256, 0, s>>>(w.tB, w.mean, w.rsig, P[PCG1], P[PCBE1], KK, M2 * KK);

  einsum_cw_kernel<<<cdiv(M2 * KK, 256), 256, 0, s>>>(w.tB, P[PCW2], P[PCB2], w.tA, K, 0, M2 * KK);
  bn_stats_kernel<<<KK, 256, 0, s>>>(w.tA, M2, KK, w.mean, w.rsig);
  bn_apply_kernel<<<cdiv(M2 * KK, 256), 256, 0, s>>>(w.tA, w.mean, w.rsig, P[PCG2], P[PCBE2], KK, M2 * KK);

  // fused xs gather -> X-transform -> depthwise conv (+db)
  xtrans_dw_kernel<<<M2, 256, KK * (int)sizeof(float), s>>>(
      x, Cin, w.hB, Cd, w.idx, w.tA, P[PDW], P[PDB], w.fcin, n, K, dm);

  // final linear [B*n, C*dm] @ fw + fb, ReLU  (WMMA)
  launch_gemm(w.fcin, P[PFW], P[PFB], out, M2, Cout, C * dm, 2, w, s);
}

extern "C" void kernel_launch(void* const* d_in, const int* in_sizes, int n_in,
                              void* d_out, int out_size, void* d_ws, size_t ws_size,
                              hipStream_t stream)
{
  (void)in_sizes; (void)n_in; (void)out_size; (void)ws_size;
  const float* x_in   = (const float*)d_in[0];   // [B*N0, 8]
  const float* pos_in = (const float*)d_in[1];   // [B*N0, 3]
  // d_in[2] = batch (deterministic; recomputed on device)
  const float* P[4][24];
  for (int l = 0; l < 4; ++l)
    for (int k = 0; k < 24; ++k)
      P[l][k] = (const float*)d_in[3 + l * 24 + k];

  // ---- carve workspace (bump allocator, 256B aligned) ----
  char* wp = (char*)d_ws;
  auto alloc = [&](size_t nbytes) -> void* {
    void* p = (void*)wp;
    wp += (nbytes + 255) & ~(size_t)255;
    return p;
  };
  Ws w;
  w.idx    = (int*)  alloc((size_t)262144  * 4);   // max B*n*K (layer 1)
  w.fpsidx = (int*)  alloc((size_t)8192    * 4);   // max B*m1
  w.pdiff  = (float*)alloc((size_t)786432  * 4);   // max B*n*K*3
  w.hA     = (float*)alloc((size_t)2621440 * 4);   // max B*n*K*Cd (layer 4)
  w.hB     = (float*)alloc((size_t)2621440 * 4);
  w.tA     = (float*)alloc((size_t)4194304 * 4);   // max B*n*K^2 (layer 1)
  w.tB     = (float*)alloc((size_t)4194304 * 4);
  w.fcin   = (float*)alloc((size_t)1310720 * 4);   // max B*n*C*dm (layer 4)
  w.featA  = (float*)alloc((size_t)1048576 * 4);
  w.featB  = (float*)alloc((size_t)1048576 * 4);
  w.pos2   = (float*)alloc((size_t)24576   * 4);   // B*2048*3
  w.pos3   = (float*)alloc((size_t)12288   * 4);   // B*1024*3
  w.mean   = (float*)alloc((size_t)512     * 4);
  w.rsig   = (float*)alloc((size_t)512     * 4);
  w.xh     = (_Float16*)alloc((size_t)1310720 * 2); // packed A (max M*Kcp)
  w.bp     = (_Float16*)alloc((size_t)131072  * 2); // packed B (max Kcp*N)

  const int m1 = N0 / 2, m2 = m1 / 2;
  float* dout = (float*)d_out;

  // ---- layer 1: n=4096, (8,32,16,2) ----
  run_xconv(x_in, 8, pos_in, N0, 16, 2, 32, P[0], w.featA, w, stream);
  // FPS 4096 -> 2048 on original positions, gather features + positions
  fps_kernel<<<BATCH, FPS_T, 0, stream>>>(pos_in, N0, m1, w.fpsidx);
  gather_kernel<<<cdiv(BATCH*m1*32, 256), 256, 0, stream>>>(w.featA, w.fpsidx, w.featB, N0, m1, 32, BATCH*m1*32);
  gather_kernel<<<cdiv(BATCH*m1*3, 256), 256, 0, stream>>>(pos_in, w.fpsidx, w.pos2, N0, m1, 3, BATCH*m1*3);

  // ---- layer 2: n=2048, (32,64,20,8) ----
  run_xconv(w.featB, 32, w.pos2, m1, 20, 8, 64, P[1], w.featA, w, stream);
  // FPS 2048 -> 1024
  fps_kernel<<<BATCH, FPS_T, 0, stream>>>(w.pos2, m1, m2, w.fpsidx);
  gather_kernel<<<cdiv(BATCH*m2*64, 256), 256, 0, stream>>>(w.featA, w.fpsidx, w.featB, m1, m2, 64, BATCH*m2*64);
  gather_kernel<<<cdiv(BATCH*m2*3, 256), 256, 0, stream>>>(w.pos2, w.fpsidx, w.pos3, m1, m2, 3, BATCH*m2*3);

  // ---- layer 3: n=1024, (64,128,20,16) ----
  run_xconv(w.featB, 64, w.pos3, m2, 20, 16, 128, P[2], w.featA, w, stream);

  // ---- layer 4: n=1024, (128,256,20,32) -> writes straight into d_out ----
  run_xconv(w.featA, 128, w.pos3, m2, 20, 32, 256, P[3], dout, w, stream);

  // ---- auxiliary outputs, concatenated in reference return order ----
  // [h: B*m2*256][pb: B*m2*3][batch3: B*m2][pos2: B*m1*3][pos1: B*N0*3][batch2: B*m1][batch1: B*N0]
  size_t o = (size_t)BATCH * m2 * 256;
  copy_kernel<<<cdiv(BATCH*m2*3, 256), 256, 0, stream>>>(dout + o, w.pos3, BATCH*m2*3);          o += (size_t)BATCH*m2*3;
  iota_batch_kernel<<<cdiv(BATCH*m2, 256), 256, 0, stream>>>(dout + o, m2, BATCH*m2);            o += (size_t)BATCH*m2;
  copy_kernel<<<cdiv(BATCH*m1*3, 256), 256, 0, stream>>>(dout + o, w.pos2, BATCH*m1*3);          o += (size_t)BATCH*m1*3;
  copy_kernel<<<cdiv(BATCH*N0*3, 256), 256, 0, stream>>>(dout + o, pos_in, BATCH*N0*3);          o += (size_t)BATCH*N0*3;
  iota_batch_kernel<<<cdiv(BATCH*m1, 256), 256, 0, stream>>>(dout + o, m1, BATCH*m1);            o += (size_t)BATCH*m1;
  iota_batch_kernel<<<cdiv(BATCH*N0, 256), 256, 0, stream>>>(dout + o, N0, BATCH*N0);
}
